// ExplicitRuleInference_35304631174166
// MI455X (gfx1250) — compile-verified, matrix operand order checked by reading, not run on hardware
//
#include <hip/hip_runtime.h>
#include <hip/hip_bf16.h>

typedef __attribute__((ext_vector_type(16))) __bf16 v16bf;
typedef __attribute__((ext_vector_type(8)))  __bf16 v8bf;
typedef __attribute__((ext_vector_type(8)))  float  v8f;
typedef __attribute__((ext_vector_type(4)))  float  v4f;

#define WMMA_BF16(a, b, c) \
  __builtin_amdgcn_wmma_f32_16x16x32_bf16(false, (a), false, (b), (short)0, (c), false, false)

// ======================= fragment helpers (ISA 7.12.2 wave32 layouts) =======================
// A 16x32 (MxK), bf16 row-major source. lanes 0-15: row=lane, K=k0+0..7 / +16..23 ; lanes 16-31: K +8.
__device__ __forceinline__ v16bf load_a_bf(const __bf16* __restrict__ A, int lda, int m0, int k0, int lane) {
  const int r  = m0 + (lane & 15);
  const int kb = k0 + ((lane & 16) ? 8 : 0);
  const __bf16* p = A + (size_t)r * lda + kb;   // 16B aligned (kb % 8 == 0, lda % 8 == 0)
  const v8bf lo = *(const v8bf*)(p);
  const v8bf hi = *(const v8bf*)(p + 16);
  v16bf a;
#pragma unroll
  for (int i = 0; i < 8; ++i) { a[i] = lo[i]; a[i + 8] = hi[i]; }
  return a;
}

// B fragment from pre-packed buffer: tile (kt, nt) of a K x N matrix, 512 bf16 per tile,
// lane's 16 values contiguous -> 32-byte aligned vector load (2x b128).
__device__ __forceinline__ v16bf load_b_p(const __bf16* __restrict__ Bp, int Ktiles, int kt, int nt, int lane) {
  return *(const v16bf*)(Bp + ((size_t)(nt * Ktiles + kt)) * 512 + lane * 16);
}

// C/D: 8 f32 VGPRs; lanes 0-15: col=n0+lane rows 0..7 ; lanes 16-31: rows 8..15
__device__ __forceinline__ void store_c_lds(float* __restrict__ T, int ldc, int n0, int lane, v8f c,
                                            const float* __restrict__ bias) {
  const int col = n0 + (lane & 15);
  const int rb  = (lane & 16) ? 8 : 0;
  const float bv = bias[col];
#pragma unroll
  for (int i = 0; i < 8; ++i) T[(rb + i) * ldc + col] = c[i] + bv;
}

// ======================= K0a: elementwise f32 -> bf16 =======================
__global__ __launch_bounds__(256) void cvt_bf16_kernel(const float* __restrict__ src, __bf16* __restrict__ dst,
                                                       unsigned n) {
  unsigned i = blockIdx.x * 256u + threadIdx.x;
  if (i < n) dst[i] = (__bf16)src[i];
}

// ======================= K0b: pack f32 [batch,K,N] weights -> bf16 B-fragment order =======================
__global__ __launch_bounds__(256) void pack_b_kernel(const float* __restrict__ src, __bf16* __restrict__ dst,
                                                     int K, int N, int batch) {
  const size_t elems = (size_t)K * N;
  const size_t total = elems * batch;
  size_t idx = (size_t)blockIdx.x * 256 + threadIdx.x;
  if (idx >= total) return;
  const size_t b  = idx / elems;
  const size_t r0 = idx % elems;
  const int Ktiles = K >> 5;
  const int tile   = (int)(r0 >> 9);
  const int within = (int)(r0 & 511);
  const int lane = within >> 4, i = within & 15;
  const int nt = tile / Ktiles, kt = tile % Ktiles;
  const int k = kt * 32 + ((lane & 16) ? 8 : 0) + (i < 8 ? i : i + 8);
  const int c = nt * 16 + (lane & 15);
  dst[idx] = (__bf16)src[b * elems + (size_t)k * N + c];
}

// ======================= K1: h = relu(LN(X @ W1 + b1)) ; [8192,128]x[128,256] -> bf16 =======================
__global__ __launch_bounds__(256) void rp1_kernel(const __bf16* __restrict__ Xb, const __bf16* __restrict__ W1p,
                                                  const float* __restrict__ b1, const float* __restrict__ g1,
                                                  const float* __restrict__ be1, __bf16* __restrict__ H) {
  __shared__ float tile[16 * 256];
  __shared__ float mu[16], rs[16];
  const int m0 = blockIdx.x * 16;
  const int wave = threadIdx.x >> 5, lane = threadIdx.x & 31;
  const int nt = wave * 2;
  v8f acc0 = {}, acc1 = {};
#pragma unroll
  for (int kt = 0; kt < 4; ++kt) {
    v16bf a = load_a_bf(Xb, 128, m0, kt * 32, lane);
    acc0 = WMMA_BF16(a, load_b_p(W1p, 4, kt, nt, lane), acc0);
    acc1 = WMMA_BF16(a, load_b_p(W1p, 4, kt, nt + 1, lane), acc1);
  }
  store_c_lds(tile, 256, nt * 16, lane, acc0, b1);
  store_c_lds(tile, 256, nt * 16 + 16, lane, acc1, b1);
  __syncthreads();
  if (threadIdx.x < 16) {
    const float* r = tile + threadIdx.x * 256;
    float s = 0.f, s2 = 0.f;
    for (int j = 0; j < 256; ++j) { float x = r[j]; s += x; s2 += x * x; }
    float m = s * (1.f / 256.f);
    float v = s2 * (1.f / 256.f) - m * m;
    mu[threadIdx.x] = m; rs[threadIdx.x] = rsqrtf(v + 1e-5f);
  }
  __syncthreads();
  {
    const int row = threadIdx.x >> 4, c0 = (threadIdx.x & 15) * 16;
    const float m = mu[row], r = rs[row];
    __bf16* o = H + (size_t)(m0 + row) * 256 + c0;
    v8bf o0, o1;
#pragma unroll
    for (int j = 0; j < 8; ++j) {
      float y0 = (tile[row * 256 + c0 + j] - m) * r * g1[c0 + j] + be1[c0 + j];
      float y1 = (tile[row * 256 + c0 + 8 + j] - m) * r * g1[c0 + 8 + j] + be1[c0 + 8 + j];
      o0[j] = (__bf16)(y0 > 0.f ? y0 : 0.f);
      o1[j] = (__bf16)(y1 > 0.f ? y1 : 0.f);
    }
    *(v8bf*)(o) = o0;
    *(v8bf*)(o + 8) = o1;
  }
}

// ======================= K2: features = relu(LN(h @ W2 + b2)) ; [8192,256]x[256,512] -> bf16 =======================
__global__ __launch_bounds__(256) void rp2_kernel(const __bf16* __restrict__ H, const __bf16* __restrict__ W2p,
                                                  const float* __restrict__ b2, const float* __restrict__ g2,
                                                  const float* __restrict__ be2, __bf16* __restrict__ F) {
  __shared__ float tile[16 * 512];
  __shared__ float mu[16], rs[16];
  const int m0 = blockIdx.x * 16;
  const int wave = threadIdx.x >> 5, lane = threadIdx.x & 31;
  const int nt = wave * 4;
  v8f acc[4] = {};
#pragma unroll 4
  for (int kt = 0; kt < 8; ++kt) {
    v16bf a = load_a_bf(H, 256, m0, kt * 32, lane);
#pragma unroll
    for (int t = 0; t < 4; ++t)
      acc[t] = WMMA_BF16(a, load_b_p(W2p, 8, kt, nt + t, lane), acc[t]);
  }
#pragma unroll
  for (int t = 0; t < 4; ++t) store_c_lds(tile, 512, (nt + t) * 16, lane, acc[t], b2);
  __syncthreads();
  if (threadIdx.x < 16) {
    const float* r = tile + threadIdx.x * 512;
    float s = 0.f, s2 = 0.f;
    for (int j = 0; j < 512; ++j) { float x = r[j]; s += x; s2 += x * x; }
    float m = s * (1.f / 512.f);
    float v = s2 * (1.f / 512.f) - m * m;
    mu[threadIdx.x] = m; rs[threadIdx.x] = rsqrtf(v + 1e-5f);
  }
  __syncthreads();
  {
    const int row = threadIdx.x >> 4, c0 = (threadIdx.x & 15) * 32;
    const float m = mu[row], r = rs[row];
    __bf16* o = F + (size_t)(m0 + row) * 512 + c0;
#pragma unroll
    for (int q = 0; q < 4; ++q) {
      v8bf ov;
#pragma unroll
      for (int j = 0; j < 8; ++j) {
        int c = c0 + q * 8 + j;
        float y = (tile[row * 512 + c] - m) * r * g2[c] + be2[c];
        ov[j] = (__bf16)(y > 0.f ? y : 0.f);
      }
      *(v8bf*)(o + q * 8) = ov;
    }
  }
}

// ======================= K3: logits = F @ op_w + op_b ; softmax ; top-3 =======================
__global__ __launch_bounds__(128) void op_topk_kernel(const __bf16* __restrict__ F, const __bf16* __restrict__ OPWp,
                                                      const float* __restrict__ OPB,
                                                      float* __restrict__ out_probs, float* __restrict__ out_idxf,
                                                      int* __restrict__ ws_idx, float* __restrict__ ws_prob) {
  __shared__ float lt[16 * 64];
  const int m0 = blockIdx.x * 16;
  const int wave = threadIdx.x >> 5, lane = threadIdx.x & 31;
  v8f acc = {};
#pragma unroll 4
  for (int kt = 0; kt < 16; ++kt) {
    v16bf a = load_a_bf(F, 512, m0, kt * 32, lane);
    acc = WMMA_BF16(a, load_b_p(OPWp, 16, kt, wave, lane), acc);
  }
  store_c_lds(lt, 64, wave * 16, lane, acc, OPB);
  __syncthreads();
  if (threadIdx.x < 16) {
    const float* r = lt + threadIdx.x * 64;
    float mx = r[0];
    for (int j = 1; j < 64; ++j) mx = r[j] > mx ? r[j] : mx;
    float s = 0.f;
    for (int j = 0; j < 64; ++j) s += __expf(r[j] - mx);
    const float inv = 1.f / s;
    const int row = m0 + threadIdx.x;
    int c0 = -1, c1 = -1;
#pragma unroll
    for (int k = 0; k < 3; ++k) {
      int bi = -1; float bv = -3.4e38f;
      for (int j = 0; j < 64; ++j) {
        if (j == c0 || j == c1) continue;
        if (r[j] > bv) { bv = r[j]; bi = j; }
      }
      const float prob = __expf(bv - mx) * inv;
      out_probs[row * 3 + k] = prob;
      out_idxf[row * 3 + k]  = (float)bi;
      ws_idx[row * 3 + k]    = bi;
      ws_prob[row * 3 + k]   = prob;
      if (k == 0) c0 = bi; else if (k == 1) c1 = bi;
    }
  }
}

// ======================= K4: per-expert FFN on selected row tiles, masked scatter =======================
__global__ __launch_bounds__(256) void expert_kernel(const __bf16* __restrict__ F, const __bf16* __restrict__ PW1p,
                                                     const float* __restrict__ PB1, const float* __restrict__ PG,
                                                     const float* __restrict__ PBE, const __bf16* __restrict__ PW2p,
                                                     const float* __restrict__ PB2, const int* __restrict__ ws_idx,
                                                     const float* __restrict__ ws_prob, float* __restrict__ outP) {
  __shared__ __align__(32) __bf16 Atile[16 * 512];
  __shared__ float  ctile[16 * 256];
  __shared__ __align__(32) __bf16 ztile[16 * 256];
  __shared__ float  out16[16 * 16];
  __shared__ int selk[16];
  __shared__ int anyflag;
  __shared__ float mu[16], rs[16];
  const int m0 = blockIdx.x * 16;
  const int e  = blockIdx.y;
  if (threadIdx.x < 16) {
    const int row = m0 + threadIdx.x;
    int sel = -1;
#pragma unroll
    for (int k = 0; k < 3; ++k)
      if (ws_idx[row * 3 + k] == e) sel = k;
    selk[threadIdx.x] = sel;
  }
  if (threadIdx.x == 0) anyflag = 0;
  __syncthreads();
  if (threadIdx.x == 0) {
    int any = 0;
    for (int t = 0; t < 16; ++t) any |= (selk[t] >= 0) ? 1 : 0;
    anyflag = any;
  }
  __syncthreads();
  if (!anyflag) return;  // expert unused by this row tile: skip all compute

  // stage contiguous bf16 feature tile into LDS (16x512 = 16KB), vectorized copy
  {
    const v8bf* s = (const v8bf*)(F + (size_t)m0 * 512);
    v8bf* d = (v8bf*)Atile;
    for (int i = threadIdx.x; i < 16 * 512 / 8; i += 256) d[i] = s[i];
  }
  __syncthreads();

  const int wave = threadIdx.x >> 5, lane = threadIdx.x & 31;
  const __bf16* W1p = PW1p + (size_t)e * (16 * 16 * 512);  // Ktiles=16, Ntiles=16
  const float*  B1  = PB1 + (size_t)e * 256;
  const int nt = wave * 2;
  v8f acc0 = {}, acc1 = {};
#pragma unroll 4
  for (int kt = 0; kt < 16; ++kt) {
    v16bf a = load_a_bf(Atile, 512, 0, kt * 32, lane);  // LDS source -> ds_load_b128
    acc0 = WMMA_BF16(a, load_b_p(W1p, 16, kt, nt, lane), acc0);
    acc1 = WMMA_BF16(a, load_b_p(W1p, 16, kt, nt + 1, lane), acc1);
  }
  store_c_lds(ctile, 256, nt * 16, lane, acc0, B1);
  store_c_lds(ctile, 256, nt * 16 + 16, lane, acc1, B1);
  __syncthreads();
  if (threadIdx.x < 16) {
    const float* r = ctile + threadIdx.x * 256;
    float s = 0.f, s2 = 0.f;
    for (int j = 0; j < 256; ++j) { float x = r[j]; s += x; s2 += x * x; }
    float m = s * (1.f / 256.f);
    float v = s2 * (1.f / 256.f) - m * m;
    mu[threadIdx.x] = m; rs[threadIdx.x] = rsqrtf(v + 1e-5f);
  }
  __syncthreads();
  {
    const float* G  = PG  + (size_t)e * 256;
    const float* BE = PBE + (size_t)e * 256;
    const int row = threadIdx.x >> 4, c0 = (threadIdx.x & 15) * 16;
    const float m = mu[row], r = rs[row];
    v8bf z0, z1;
#pragma unroll
    for (int j = 0; j < 8; ++j) {
      float y0 = (ctile[row * 256 + c0 + j] - m) * r * G[c0 + j] + BE[c0 + j];
      float y1 = (ctile[row * 256 + c0 + 8 + j] - m) * r * G[c0 + 8 + j] + BE[c0 + 8 + j];
      z0[j] = (__bf16)(y0 > 0.f ? y0 : 0.f);
      z1[j] = (__bf16)(y1 > 0.f ? y1 : 0.f);
    }
    *(v8bf*)(ztile + row * 256 + c0) = z0;
    *(v8bf*)(ztile + row * 256 + c0 + 8) = z1;
  }
  __syncthreads();
  if (wave == 0) {  // 16x16 output tile: K=256 -> 8 wmma
    const __bf16* W2p = PW2p + (size_t)e * (8 * 512);  // Ktiles=8, Ntiles=1
    const float*  B2  = PB2 + (size_t)e * 16;
    v8f acc = {};
#pragma unroll 4
    for (int kt = 0; kt < 8; ++kt) {
      v16bf a = load_a_bf(ztile, 256, 0, kt * 32, lane);
      acc = WMMA_BF16(a, load_b_p(W2p, 8, kt, 0, lane), acc);
    }
    store_c_lds(out16, 16, 0, lane, acc, B2);
  }
  __syncthreads();
  if (threadIdx.x < 16) {
    const int sel = selk[threadIdx.x];
    if (sel >= 0) {
      const int row = m0 + threadIdx.x;
      const float prob = ws_prob[row * 3 + sel];
      const float msk = (prob >= 0.1f) ? 1.f : 0.f;
      v4f* o4 = (v4f*)(outP + ((size_t)row * 3 + sel) * 16);
#pragma unroll
      for (int q = 0; q < 4; ++q) {
        v4f v;
#pragma unroll
        for (int j = 0; j < 4; ++j) v[j] = out16[threadIdx.x * 16 + q * 4 + j] * msk;
        o4[q] = v;
      }
    }
  }
}

extern "C" void kernel_launch(void* const* d_in, const int* in_sizes, int n_in,
                              void* d_out, int out_size, void* d_ws, size_t ws_size,
                              hipStream_t stream) {
  (void)in_sizes; (void)n_in; (void)out_size; (void)ws_size;
  const int B = 8192;
  const float* X    = (const float*)d_in[0];
  const float* W1   = (const float*)d_in[1];
  const float* b1   = (const float*)d_in[2];
  const float* g1   = (const float*)d_in[3];
  const float* be1  = (const float*)d_in[4];
  const float* W2   = (const float*)d_in[5];
  const float* b2   = (const float*)d_in[6];
  const float* g2   = (const float*)d_in[7];
  const float* be2  = (const float*)d_in[8];
  const float* OPW  = (const float*)d_in[9];
  const float* OPB  = (const float*)d_in[10];
  const float* PW1  = (const float*)d_in[11];
  const float* PB1  = (const float*)d_in[12];
  const float* PG   = (const float*)d_in[13];
  const float* PBE  = (const float*)d_in[14];
  const float* PW2  = (const float*)d_in[15];
  const float* PB2  = (const float*)d_in[16];

  // ---- workspace layout ----
  float* wsp = (float*)d_ws;                       // [B,3] probs
  int*   wsi = (int*)(wsp + (size_t)B * 3);        // [B,3] idx
  __bf16* Xb   = (__bf16*)(wsi + (size_t)B * 3);   // [B,128]
  __bf16* Hb   = Xb + (size_t)B * 128;             // [B,256]
  __bf16* Fb   = Hb + (size_t)B * 256;             // [B,512]
  __bf16* W1p  = Fb + (size_t)B * 512;             // 128x256 packed  (32768)
  __bf16* W2p  = W1p + 32768;                      // 256x512 packed  (131072)
  __bf16* OPWp = W2p + 131072;                     // 512x64 packed   (32768)
  __bf16* PW1p = OPWp + 32768;                     // 64 x (512x256)  (8388608)
  __bf16* PW2p = PW1p + (size_t)64 * 131072;       // 64 x (256x16)   (262144)

  float* out        = (float*)d_out;
  float* out_probs  = out;                  // [8192,3]
  float* out_idxf   = out + (size_t)B * 3;  // [8192,3]
  float* out_params = out + (size_t)B * 6;  // [8192,3,16]

  // convert activation input + pack weights into bf16 WMMA-B fragment order
  cvt_bf16_kernel<<<(B * 128 + 255) / 256, 256, 0, stream>>>(X, Xb, B * 128);
  pack_b_kernel<<<(32768 + 255) / 256, 256, 0, stream>>>(W1, W1p, 128, 256, 1);
  pack_b_kernel<<<(131072 + 255) / 256, 256, 0, stream>>>(W2, W2p, 256, 512, 1);
  pack_b_kernel<<<(32768 + 255) / 256, 256, 0, stream>>>(OPW, OPWp, 512, 64, 1);
  pack_b_kernel<<<(8388608 + 255) / 256, 256, 0, stream>>>(PW1, PW1p, 512, 256, 64);
  pack_b_kernel<<<(262144 + 255) / 256, 256, 0, stream>>>(PW2, PW2p, 256, 16, 64);

  rp1_kernel<<<512, 256, 0, stream>>>(Xb, W1p, b1, g1, be1, Hb);
  rp2_kernel<<<512, 256, 0, stream>>>(Hb, W2p, b2, g2, be2, Fb);
  op_topk_kernel<<<512, 128, 0, stream>>>(Fb, OPWp, OPB, out_probs, out_idxf, wsi, wsp);
  expert_kernel<<<dim3(512, 64), 256, 0, stream>>>(Fb, PW1p, PB1, PG, PBE, PW2p, PB2, wsi, wsp, out_params);
}
